// SpikingPCNetwork_30116310680260
// MI455X (gfx1250) — compile-verified, matrix operand order checked by reading, not run on hardware
//
#include <hip/hip_runtime.h>

typedef __attribute__((ext_vector_type(16))) _Float16 v16h;
typedef __attribute__((ext_vector_type(8)))  float    v8f;

// ---------------------------------------------------------------------------
// Kernel 1: s0[r] = sum_k W0[r,k]  (r < 1024, K = 512)
//           s1[r] = sum_k W1[r,k]  (r < 512,  K = 256)
// computed as D = A x B with B == ones via v_wmma_f32_16x16x32_f16.
// One wave32 per 16-row tile. blocks 0..63 -> W0, 64..95 -> W1.
// ---------------------------------------------------------------------------
__global__ __launch_bounds__(32) void rowsum_wmma_kernel(
    const float* __restrict__ W0, const float* __restrict__ W1,
    float* __restrict__ s /* [0,1024) = s0 ; [1024,1536) = s1 */) {
  const int lane = threadIdx.x & 31;
  const int hi   = lane >> 4;      // 0: K 0-7/16-23, 1: K 8-15/24-31
  const int mrow = lane & 15;      // A-matrix row M within tile

  const float* W;
  float* out;
  int cols, row0;
  if ((int)blockIdx.x < 64) {
    W = W0; cols = 512; out = s;        row0 = blockIdx.x * 16;
  } else {
    W = W1; cols = 256; out = s + 1024; row0 = (blockIdx.x - 64) * 16;
  }
  const int r = row0 + mrow;
  const float* wrow = W + (size_t)r * cols;

  v16h bmat;
#pragma unroll
  for (int i = 0; i < 16; ++i) bmat[i] = (_Float16)1.0f;

  v8f c = {};
  for (int kc = 0; kc < cols; kc += 32) {
    const int k0 = kc + hi * 8;
    float4 p0 = *(const float4*)(wrow + k0);
    float4 p1 = *(const float4*)(wrow + k0 + 4);
    float4 p2 = *(const float4*)(wrow + k0 + 16);
    float4 p3 = *(const float4*)(wrow + k0 + 20);
    v16h a;
    a[0]  = (_Float16)p0.x; a[1]  = (_Float16)p0.y;
    a[2]  = (_Float16)p0.z; a[3]  = (_Float16)p0.w;
    a[4]  = (_Float16)p1.x; a[5]  = (_Float16)p1.y;
    a[6]  = (_Float16)p1.z; a[7]  = (_Float16)p1.w;
    a[8]  = (_Float16)p2.x; a[9]  = (_Float16)p2.y;
    a[10] = (_Float16)p2.z; a[11] = (_Float16)p2.w;
    a[12] = (_Float16)p3.x; a[13] = (_Float16)p3.y;
    a[14] = (_Float16)p3.z; a[15] = (_Float16)p3.w;
    // D[m,n] = sum_k A[m,k] * 1 + C[m,n]  (all n columns identical)
    c = __builtin_amdgcn_wmma_f32_16x16x32_f16(
        /*neg_a=*/false, a, /*neg_b=*/false, bmat,
        /*c_mod=*/(short)0, c, /*reuse_a=*/false, /*reuse_b=*/false);
  }

  // D layout: lane 0 -> M = 0..7 (VGPR i, N=0); lane 16 -> M = 8..15.
  if (lane == 0) {
#pragma unroll
    for (int i = 0; i < 8; ++i) out[row0 + i] = c[i];
  } else if (lane == 16) {
#pragma unroll
    for (int i = 0; i < 8; ++i) out[row0 + 8 + i] = c[i];
  }
}

// ---------------------------------------------------------------------------
// Kernel 2: collapsed SNN recurrence. One wave32 per batch row.
// x row (1024 f32) + s0 (1024) live in VGPRs (32 per lane), s1 (512) too.
// Per step: |x - a0*s0| reduction (butterfly), scalar LIF, |a0 - a1*s1|
// reduction, scalar LIF. Output: a1 broadcast over 256 columns.
// ---------------------------------------------------------------------------
__global__ __launch_bounds__(256) void snn_lif_kernel(
    const float* __restrict__ x, const float* __restrict__ s,
    const int* __restrict__ Tptr, float* __restrict__ out) {
  const int lane = threadIdx.x & 31;
  const int wave = threadIdx.x >> 5;
  const int b    = blockIdx.x * 8 + wave;
  const int T    = *Tptr;

  const float* xb = x + (size_t)b * 1024;
  const float* s0 = s;
  const float* s1 = s + 1024;

  float xr[32], s0r[32], s1r[16];
#pragma unroll
  for (int c = 0; c < 8; ++c) {
    float4 xv = *(const float4*)(xb + c * 128 + lane * 4);
    float4 sv = *(const float4*)(s0 + c * 128 + lane * 4);
    xr[4 * c + 0] = xv.x; xr[4 * c + 1] = xv.y;
    xr[4 * c + 2] = xv.z; xr[4 * c + 3] = xv.w;
    s0r[4 * c + 0] = sv.x; s0r[4 * c + 1] = sv.y;
    s0r[4 * c + 2] = sv.z; s0r[4 * c + 3] = sv.w;
  }
#pragma unroll
  for (int c = 0; c < 4; ++c) {
    float4 sv = *(const float4*)(s1 + c * 128 + lane * 4);
    s1r[4 * c + 0] = sv.x; s1r[4 * c + 1] = sv.y;
    s1r[4 * c + 2] = sv.z; s1r[4 * c + 3] = sv.w;
  }

  float v0 = 0.f, a0 = 0.f, v1 = 0.f, a1 = 0.f;

  for (int t = 0; t < T; ++t) {
    const float tf = (float)t;

    // ---- layer 0: cur0 = (2/1024) * sum_j |x[j] - a0*s0[j]| ----
    float acc = 0.f;
#pragma unroll
    for (int k = 0; k < 32; ++k)
      acc += fabsf(fmaf(-a0, s0r[k], xr[k]));   // v_fma + v_add |src|
#pragma unroll
    for (int off = 16; off > 0; off >>= 1)
      acc += __shfl_xor(acc, off, 32);          // wave32 butterfly (ds path)
    const float cur0 = acc * (2.0f / 1024.0f);
    v0 = 0.5f * v0 + cur0;
    const float sp0 = (v0 >= 1.0f) ? 1.0f : 0.0f;
    v0 *= (1.0f - sp0);
    a0 = (a0 * tf + sp0) / (tf + 1.0f);

    // ---- layer 1: cur1 = (2/512) * sum_j |a0 - a1*s1[j]| ----
    float acc1 = 0.f;
#pragma unroll
    for (int k = 0; k < 16; ++k)
      acc1 += fabsf(fmaf(-a1, s1r[k], a0));
#pragma unroll
    for (int off = 16; off > 0; off >>= 1)
      acc1 += __shfl_xor(acc1, off, 32);
    const float cur1 = acc1 * (2.0f / 512.0f);
    v1 = 0.5f * v1 + cur1;
    const float sp1 = (v1 >= 1.0f) ? 1.0f : 0.0f;
    v1 *= (1.0f - sp1);
    a1 = (a1 * tf + sp1) / (tf + 1.0f);
  }

  // out[b, 0..255] = a1 (column-constant top-layer avg)
  float* ob = out + (size_t)b * 256;
  float4 o; o.x = o.y = o.z = o.w = a1;
  *(float4*)(ob + lane * 4)       = o;
  *(float4*)(ob + 128 + lane * 4) = o;
}

// ---------------------------------------------------------------------------
extern "C" void kernel_launch(void* const* d_in, const int* in_sizes, int n_in,
                              void* d_out, int out_size, void* d_ws, size_t ws_size,
                              hipStream_t stream) {
  const float* x  = (const float*)d_in[0];   // [32768, 1024]
  const float* W0 = (const float*)d_in[1];   // [1024, 512]
  const float* W1 = (const float*)d_in[2];   // [512, 256]
  const int*   T  = (const int*)d_in[3];     // time_steps (device scalar)
  float* s = (float*)d_ws;                   // 1536 floats: s0 | s1

  // 64 tiles for W0 (1024 rows) + 32 tiles for W1 (512 rows), one wave each.
  rowsum_wmma_kernel<<<96, 32, 0, stream>>>(W0, W1, s);

  // 32768 rows, one wave32 per row, 8 waves per block.
  snn_lif_kernel<<<32768 / 8, 256, 0, stream>>>(x, s, T, (float*)d_out);
}
